// PhysicsAttentionIrregular_6966436954794
// MI455X (gfx1250) — compile-verified, matrix operand order checked by reading, not run on hardware
//
#include <hip/hip_runtime.h>
#include <hip/hip_bf16.h>

// ---------------------------------------------------------------------------
// PhysicsAttentionIrregular for MI455X (gfx1250, wave32, WMMA bf16).
// Big GEMMs: TDM (tensor_load_to_lds) double-buffered LDS staging feeding
// v_wmma_f32_16x16x32_bf16 (f32 accumulate). Tiny stages stay f32 VALU.
// ---------------------------------------------------------------------------

typedef __attribute__((ext_vector_type(16))) __bf16 bf16x16;
typedef __attribute__((ext_vector_type(8)))  float  f32x8;

#define DIMC   256
#define HEADS  8
#define DHEAD  64
#define GSLICE 64
#define INNER  512
#define BB     4
#define NTOK   32768

#if defined(__gfx1250__) && __has_builtin(__builtin_amdgcn_tensor_load_to_lds)
#define USE_TDM 1
#else
#define USE_TDM 0
#endif

__device__ __forceinline__ unsigned short f32_to_bf16(float f) {
  unsigned u = __float_as_uint(f);
  unsigned r = u + 0x7fffu + ((u >> 16) & 1u);   // round-to-nearest-even
  return (unsigned short)(r >> 16);
}
__device__ __forceinline__ float bf16_to_f32(unsigned short h) {
  return __uint_as_float(((unsigned)h) << 16);
}
__device__ __forceinline__ f32x8 wmma_bf16(bf16x16 a, bf16x16 b, f32x8 c) {
  return __builtin_amdgcn_wmma_f32_16x16x32_bf16(false, a, false, b, (short)0, c,
                                                 false, false);
}

#if USE_TDM
typedef __attribute__((ext_vector_type(4))) unsigned int u32x4;
typedef __attribute__((ext_vector_type(8))) int          i32x8;
typedef __attribute__((ext_vector_type(4))) int          i32x4;

// Issue one TDM 2-D tile load: tile_d1 rows x 32 bf16 elements, row stride
// `row_stride` elements, into LDS at byte offset lds_addr with 4 DWORDs of
// padding after every 16 DWORDs (-> 40-half LDS row stride, matches the
// conflict-free fragment layout). D# bit packing per CDNA5 ISA ch. 8.
// This toolchain exposes the 6-arg builtin (extra i32x8 group before cpol).
__device__ __forceinline__ void tdm_load_tile32(unsigned lds_addr,
                                                const void* gaddr,
                                                unsigned tile_d1,
                                                unsigned row_stride) {
  unsigned long long ga = (unsigned long long)gaddr;
  u32x4 g0;
  g0[0] = 1u;  // count=1 (valid user descriptor)
  g0[1] = lds_addr;
  g0[2] = (unsigned)(ga & 0xffffffffu);
  g0[3] = (unsigned)((ga >> 32) & 0x1ffffffu) | (2u << 30);  // type=2 ("image")
  const unsigned td0 = 0x40000000u, td1 = 0x40000000u;  // huge dims: no OOB
  const unsigned tile_d0 = 32u;
  i32x8 g1;
  // data_size=1 (2B) | pad_enable | pad_interval=3 (16 DW) | pad_amount=3 (4 DW)
  g1[0] = (int)((1u << 16) | (1u << 20) | (3u << 22) | (3u << 25));
  g1[1] = (int)((td0 & 0xffffu) << 16);
  g1[2] = (int)(((td0 >> 16) & 0xffffu) | ((td1 & 0xffffu) << 16));
  g1[3] = (int)(((td1 >> 16) & 0xffffu) | (tile_d0 << 16));
  g1[4] = (int)(tile_d1 & 0xffffu);  // tile_dim2 = 0
  g1[5] = (int)row_stride;           // tensor_dim0_stride[31:0]
  g1[6] = 0;                         // stride hi + tensor_dim1_stride lo
  g1[7] = 0;
  i32x4 z4 = {0, 0, 0, 0};
  i32x8 z8 = {0, 0, 0, 0, 0, 0, 0, 0};
  __builtin_amdgcn_tensor_load_to_lds(g0, g1, z4, z4, z8, 0);
}
__device__ __forceinline__ unsigned lds_off_u32(const void* p) {
  // generic shared address: low 32 bits are the LDS byte offset (ISA 10.2)
  return (unsigned)(unsigned long long)(uintptr_t)p;
}
#endif

// ------------------------------- utilities ---------------------------------

__global__ void cvt_f32_to_bf16_kernel(const float* __restrict__ s,
                                       unsigned short* __restrict__ d, int n) {
  for (int i = blockIdx.x * blockDim.x + threadIdx.x; i < n;
       i += gridDim.x * blockDim.x)
    d[i] = f32_to_bf16(s[i]);
}

__global__ void zero_f32_kernel(float* __restrict__ p, int n) {
  for (int i = blockIdx.x * blockDim.x + threadIdx.x; i < n;
       i += gridDim.x * blockDim.x)
    p[i] = 0.0f;
}

// ------------------------- generic bf16 WMMA GEMM --------------------------
// C[M,Nout] = A[M,K](bf16,row-major,lda) @ W[Nout,K]^T (bf16) + bias(f32)
// Block tile 128(M) x 64(N), BK=32. 8 waves; wave w owns M-strip [16w,16w+16)
// and four 16-col tiles (4 accumulators). LDS row stride = 40 halves
// (80B = 5*16B: keeps b128 alignment, kills the 16-bank stride conflict).
// Staging: wave 0 issues TDM descriptors (TENSORcnt), double buffered so the
// DMA streams slab i+1 while all waves run WMMAs on slab i.

template <int OUT_BF16>
__global__ void gemm_bf16_kernel(const unsigned short* __restrict__ A, int lda,
                                 const unsigned short* __restrict__ Wt, int K,
                                 const float* __restrict__ bias,
                                 void* __restrict__ Cout, int ldc) {
  __shared__ __align__(32) unsigned short ldsA[2][128 * 40];
  __shared__ __align__(32) unsigned short ldsW[2][64 * 40];
  const int t = threadIdx.x, w = t >> 5, l = t & 31;
  const size_t rowBase = (size_t)blockIdx.x * 128;
  const int    colBase = blockIdx.y * 64;

  f32x8 acc[4] = {};

#if USE_TDM
  if (t < 32) {  // wave 0 drives the Tensor Data Mover
    tdm_load_tile32(lds_off_u32(&ldsA[0][0]),
                    A + rowBase * (size_t)lda, 128, (unsigned)lda);
    tdm_load_tile32(lds_off_u32(&ldsW[0][0]),
                    Wt + (size_t)colBase * K, 64, (unsigned)K);
    __builtin_amdgcn_s_wait_tensorcnt(0);
  }
  __syncthreads();
  for (int k0 = 0, it = 0; k0 < K; k0 += 32, ++it) {
    const int cur = it & 1;
    if ((k0 + 32 < K) && t < 32) {  // prefetch next slab into other buffer
      tdm_load_tile32(lds_off_u32(&ldsA[cur ^ 1][0]),
                      A + rowBase * (size_t)lda + k0 + 32, 128, (unsigned)lda);
      tdm_load_tile32(lds_off_u32(&ldsW[cur ^ 1][0]),
                      Wt + (size_t)colBase * K + k0 + 32, 64, (unsigned)K);
    }
    bf16x16 a =
        *(const bf16x16*)&ldsA[cur][(w * 16 + (l & 15)) * 40 + (l >> 4) * 16];
#pragma unroll
    for (int j = 0; j < 4; ++j) {
      bf16x16 bb =
          *(const bf16x16*)&ldsW[cur][(j * 16 + (l & 15)) * 40 + (l >> 4) * 16];
      acc[j] = wmma_bf16(a, bb, acc[j]);
    }
    if (t < 32) __builtin_amdgcn_s_wait_tensorcnt(0);
    __syncthreads();
  }
#else
  for (int k0 = 0; k0 < K; k0 += 32) {
    {  // stage A tile: 128 rows x 32 halves, 2 threads/row, 32B each
      int row = t >> 1, ch = (t & 1) * 16;
      bf16x16 va = *(const bf16x16*)&A[(rowBase + row) * (size_t)lda + k0 + ch];
      *(bf16x16*)&ldsA[0][row * 40 + ch] = va;
    }
    if (t < 128) {  // stage W tile: 64 rows x 32 halves
      int row = t >> 1, ch = (t & 1) * 16;
      bf16x16 vw = *(const bf16x16*)&Wt[(size_t)(colBase + row) * K + k0 + ch];
      *(bf16x16*)&ldsW[0][row * 40 + ch] = vw;
    }
    __syncthreads();
    bf16x16 a =
        *(const bf16x16*)&ldsA[0][(w * 16 + (l & 15)) * 40 + (l >> 4) * 16];
#pragma unroll
    for (int j = 0; j < 4; ++j) {
      bf16x16 bb =
          *(const bf16x16*)&ldsW[0][(j * 16 + (l & 15)) * 40 + (l >> 4) * 16];
      acc[j] = wmma_bf16(a, bb, acc[j]);
    }
    __syncthreads();
  }
#endif

#pragma unroll
  for (int j = 0; j < 4; ++j) {
#pragma unroll
    for (int r = 0; r < 8; ++r) {
      size_t row = rowBase + w * 16 + r + 8 * (l >> 4);
      int    col = colBase + j * 16 + (l & 15);
      float  v   = acc[j][r] + bias[col];
      if (OUT_BF16)
        ((unsigned short*)Cout)[row * (size_t)ldc + col] = f32_to_bf16(v);
      else
        ((float*)Cout)[row * (size_t)ldc + col] = v;
    }
  }
}

// ---------------- fused slice logits + softmax + norm reduce ---------------
// grid = (N/64, B*H), block = 256 (8 waves, 8 tokens each).

__global__ void slice_softmax_kernel(const unsigned short* __restrict__ xm,
                                     const float* __restrict__ Wslice,
                                     const float* __restrict__ bslice,
                                     const float* __restrict__ temp,
                                     unsigned short* __restrict__ wout,
                                     float* __restrict__ norm) {
  __shared__ float ws[64 * 65];          // +1 pad -> conflict-free column reads
  __shared__ float normAcc[64];
  __shared__ float xtok[8][8][64];
  const int t = threadIdx.x, w = t >> 5, l = t & 31;
  const int bh = blockIdx.y, b = bh >> 3, h = bh & 7;

  for (int i = t; i < 64 * 64; i += 256) ws[(i >> 6) * 65 + (i & 63)] = Wslice[i];
  if (t < 64) normAcc[t] = 0.0f;
  __syncthreads();

  const float inv_t  = 1.0f / temp[h];
  const int   n_base = blockIdx.x * 64 + w * 8;
  for (int i = 0; i < 8; ++i) {
    const unsigned short* xrow =
        xm + ((size_t)b * NTOK + n_base + i) * INNER + h * DHEAD;
    xtok[w][i][2 * l]     = bf16_to_f32(xrow[2 * l]);
    xtok[w][i][2 * l + 1] = bf16_to_f32(xrow[2 * l + 1]);
  }
  __builtin_amdgcn_wave_barrier();
  for (int i = 0; i < 8; ++i) {
    float s0 = bslice[l], s1 = bslice[l + 32];
    for (int c = 0; c < 64; ++c) {
      float xv = xtok[w][i][c];
      s0 += xv * ws[l * 65 + c];
      s1 += xv * ws[(l + 32) * 65 + c];
    }
    s0 *= inv_t; s1 *= inv_t;
    float m = fmaxf(s0, s1);
    for (int d = 16; d >= 1; d >>= 1) m = fmaxf(m, __shfl_xor(m, d, 32));
    float e0 = __expf(s0 - m), e1 = __expf(s1 - m);
    float su = e0 + e1;
    for (int d = 16; d >= 1; d >>= 1) su += __shfl_xor(su, d, 32);
    float rs = 1.0f / su;
    float w0 = e0 * rs, w1 = e1 * rs;
    size_t base = ((size_t)bh * NTOK + n_base + i) * GSLICE;
    wout[base + l]      = f32_to_bf16(w0);
    wout[base + l + 32] = f32_to_bf16(w1);
    atomicAdd(&normAcc[l], w0);
    atomicAdd(&normAcc[l + 32], w1);
  }
  __syncthreads();
  if (t < 64) atomicAdd(&norm[bh * 64 + t], normAcc[t]);
}

// -------------------- slice_token: wT @ fx  (split-K WMMA) -----------------
// grid = (16, B*H). Each block reduces chunk = N/16 rows, atomics combine.

__global__ void slice_token_kernel(const unsigned short* __restrict__ wg,
                                   const unsigned short* __restrict__ fx,
                                   float* __restrict__ st, int chunk) {
  __shared__ __align__(32) unsigned short ldsWT[64 * 40];  // [g][n] transposed
  __shared__ __align__(32) unsigned short ldsFT[64 * 40];  // [c][n] transposed
  const int t = threadIdx.x, w = t >> 5, l = t & 31;
  const int bh = blockIdx.y, b = bh >> 3, h = bh & 7;
  const int n0 = blockIdx.x * chunk;

  const int tile0 = w * 2, tile1 = w * 2 + 1;
  const int gi0 = tile0 >> 2, ci0 = tile0 & 3;
  const int gi1 = tile1 >> 2, ci1 = tile1 & 3;
  f32x8 acc0 = {}, acc1 = {};

  for (int nn = 0; nn < chunk; nn += 32) {
    const int nb = n0 + nn;
#pragma unroll
    for (int p = 0; p < 8; ++p) {  // transpose-stage 32n x 64{g,c}
      int lin = p * 256 + t;
      int g = lin & 63, nl = lin >> 6;
      ldsWT[g * 40 + nl] = wg[((size_t)bh * NTOK + nb + nl) * GSLICE + g];
      ldsFT[g * 40 + nl] =
          fx[((size_t)b * NTOK + nb + nl) * INNER + h * DHEAD + g];
    }
    __syncthreads();
    bf16x16 a0 = *(const bf16x16*)&ldsWT[(gi0 * 16 + (l & 15)) * 40 + (l >> 4) * 16];
    bf16x16 b0 = *(const bf16x16*)&ldsFT[(ci0 * 16 + (l & 15)) * 40 + (l >> 4) * 16];
    acc0 = wmma_bf16(a0, b0, acc0);
    bf16x16 a1 = *(const bf16x16*)&ldsWT[(gi1 * 16 + (l & 15)) * 40 + (l >> 4) * 16];
    bf16x16 b1 = *(const bf16x16*)&ldsFT[(ci1 * 16 + (l & 15)) * 40 + (l >> 4) * 16];
    acc1 = wmma_bf16(a1, b1, acc1);
    __syncthreads();
  }
#pragma unroll
  for (int r = 0; r < 8; ++r) {
    int g0 = gi0 * 16 + r + 8 * (l >> 4), c0 = ci0 * 16 + (l & 15);
    atomicAdd(&st[bh * 4096 + g0 * 64 + c0], acc0[r]);
    int g1 = gi1 * 16 + r + 8 * (l >> 4), c1 = ci1 * 16 + (l & 15);
    atomicAdd(&st[bh * 4096 + g1 * 64 + c1], acc1[r]);
  }
}

// ----------------- tiny per-(b,h) attention over 64 slice tokens -----------
// grid = 32 blocks, 256 threads. ~84 MFLOP total: f32 VALU, stays accurate.

__global__ void small_attn_kernel(const float* __restrict__ st,
                                  const float* __restrict__ norm,
                                  const float* __restrict__ Wq,
                                  const float* __restrict__ Wk,
                                  const float* __restrict__ Wv,
                                  unsigned short* __restrict__ os) {
  __shared__ float s_t[64 * 65];
  __shared__ float s_q[64 * 65];
  __shared__ float s_k[64 * 65];
  __shared__ float s_v[64 * 65];
  __shared__ float attnRow[8][64];
  const int t = threadIdx.x, w = t >> 5, l = t & 31;
  const int bh = blockIdx.x;
  const float scale = 0.125f;  // 64^-0.5

  for (int i = 0; i < 16; ++i) {
    int idx = t * 16 + i;
    int g = idx >> 6, c = idx & 63;
    float r = 1.0f / (norm[bh * 64 + g] + 1e-5f);
    s_t[g * 65 + c] = st[bh * 4096 + idx] * r;
  }
  __syncthreads();
  for (int i = 0; i < 16; ++i) {
    int idx = t * 16 + i;
    int g = idx >> 6, d = idx & 63;
    float aq = 0.f, ak = 0.f, av = 0.f;
    for (int c = 0; c < 64; ++c) {
      float sv = s_t[g * 65 + c];
      aq += sv * Wq[d * 64 + c];
      ak += sv * Wk[d * 64 + c];
      av += sv * Wv[d * 64 + c];
    }
    s_q[g * 65 + d] = aq;
    s_k[g * 65 + d] = ak;
    s_v[g * 65 + d] = av;
  }
  __syncthreads();
  for (int i = 0; i < 8; ++i) {
    int g = w * 8 + i;
    float s0 = 0.f, s1 = 0.f;
    for (int d = 0; d < 64; ++d) {
      float qv = s_q[g * 65 + d];
      s0 += qv * s_k[l * 65 + d];
      s1 += qv * s_k[(l + 32) * 65 + d];
    }
    s0 *= scale; s1 *= scale;
    float m = fmaxf(s0, s1);
    for (int dd = 16; dd >= 1; dd >>= 1) m = fmaxf(m, __shfl_xor(m, dd, 32));
    float e0 = __expf(s0 - m), e1 = __expf(s1 - m);
    float su = e0 + e1;
    for (int dd = 16; dd >= 1; dd >>= 1) su += __shfl_xor(su, dd, 32);
    float rs = 1.0f / su;
    attnRow[w][l]      = e0 * rs;
    attnRow[w][l + 32] = e1 * rs;
    __builtin_amdgcn_wave_barrier();
    float o0 = 0.f, o1 = 0.f;
    for (int kx = 0; kx < 64; ++kx) {
      float av = attnRow[w][kx];
      o0 += av * s_v[kx * 65 + l];
      o1 += av * s_v[kx * 65 + l + 32];
    }
    os[bh * 4096 + g * 64 + l]      = f32_to_bf16(o0);
    os[bh * 4096 + g * 64 + l + 32] = f32_to_bf16(o1);
  }
}

// -------------------- unpool: out_x = w @ out_slice (WMMA) -----------------
// grid = (N/128, B*H). K=64 (2 WMMA steps). out_slice staged transposed.

__global__ void unpool_kernel(const unsigned short* __restrict__ os,
                              const unsigned short* __restrict__ wg,
                              unsigned short* __restrict__ outx) {
  __shared__ __align__(32) unsigned short ldsOS[64 * 72];  // [c][g]
  const int t = threadIdx.x, w = t >> 5, l = t & 31;
  const int bh = blockIdx.y, b = bh >> 3, h = bh & 7;
  const size_t n0 = (size_t)blockIdx.x * 128;

#pragma unroll
  for (int p = 0; p < 16; ++p) {
    int lin = p * 256 + t;
    int c = lin & 63, g = lin >> 6;
    ldsOS[c * 72 + g] = os[bh * 4096 + g * 64 + c];
  }
  __syncthreads();

  f32x8 acc[4] = {};
#pragma unroll
  for (int kk = 0; kk < 2; ++kk) {
    int k0 = kk * 32;
    size_t nrow = (size_t)bh * NTOK + n0 + w * 16 + (l & 15);
    bf16x16 a = *(const bf16x16*)&wg[nrow * GSLICE + k0 + (l >> 4) * 16];
#pragma unroll
    for (int j = 0; j < 4; ++j) {
      bf16x16 bb =
          *(const bf16x16*)&ldsOS[(j * 16 + (l & 15)) * 72 + k0 + (l >> 4) * 16];
      acc[j] = wmma_bf16(a, bb, acc[j]);
    }
  }
#pragma unroll
  for (int j = 0; j < 4; ++j)
#pragma unroll
    for (int r = 0; r < 8; ++r) {
      size_t n = n0 + w * 16 + r + 8 * (l >> 4);
      int    c = j * 16 + (l & 15);
      outx[((size_t)b * NTOK + n) * INNER + h * DHEAD + c] =
          f32_to_bf16(acc[j][r]);
    }
}

// ------------------------------- launcher ----------------------------------

extern "C" void kernel_launch(void* const* d_in, const int* in_sizes, int n_in,
                              void* d_out, int out_size, void* d_ws,
                              size_t ws_size, hipStream_t stream) {
  (void)in_sizes; (void)n_in; (void)out_size; (void)ws_size;
  const float* x      = (const float*)d_in[0];
  const float* Wfx    = (const float*)d_in[1];
  const float* bfx    = (const float*)d_in[2];
  const float* Wx     = (const float*)d_in[3];
  const float* bx     = (const float*)d_in[4];
  const float* Wslice = (const float*)d_in[5];
  const float* bslice = (const float*)d_in[6];
  const float* temp   = (const float*)d_in[7];
  const float* Wq     = (const float*)d_in[8];
  const float* Wk     = (const float*)d_in[9];
  const float* Wv     = (const float*)d_in[10];
  const float* Wout   = (const float*)d_in[11];
  const float* bout   = (const float*)d_in[12];

  // workspace carve-up (256B aligned)
  char*  ws  = (char*)d_ws;
  size_t off = 0;
  auto carve = [&](size_t bytes) {
    size_t o = off;
    off = (off + bytes + 255) & ~(size_t)255;
    return ws + o;
  };
  const size_t M = (size_t)BB * NTOK;  // 131072 rows
  unsigned short* xb    = (unsigned short*)carve(M * DIMC * 2);
  unsigned short* fx    = (unsigned short*)carve(M * INNER * 2);
  unsigned short* xm    = (unsigned short*)carve(M * INNER * 2);  // reused: out_x
  unsigned short* wgt   = (unsigned short*)carve(M * HEADS * GSLICE * 2); // [B,H,N,G]
  float*          st    = (float*)carve((32 * 4096 + 2048) * 4);  // slice_token + norm
  float*          norm  = st + 32 * 4096;
  unsigned short* osl   = (unsigned short*)carve(32 * 4096 * 2);
  unsigned short* wfxb  = (unsigned short*)carve((size_t)INNER * DIMC * 2);
  unsigned short* wxb   = (unsigned short*)carve((size_t)INNER * DIMC * 2);
  unsigned short* woutb = (unsigned short*)carve((size_t)DIMC * INNER * 2);

  // 1) precision conversion (one pass; all big GEMMs then run bf16/WMMA)
  cvt_f32_to_bf16_kernel<<<4096, 256, 0, stream>>>(x, xb, (int)(M * DIMC));
  cvt_f32_to_bf16_kernel<<<512, 256, 0, stream>>>(Wfx, wfxb, INNER * DIMC);
  cvt_f32_to_bf16_kernel<<<512, 256, 0, stream>>>(Wx, wxb, INNER * DIMC);
  cvt_f32_to_bf16_kernel<<<512, 256, 0, stream>>>(Wout, woutb, DIMC * INNER);
  // 2) zero split-K accumulators (ws is poisoned between runs)
  zero_f32_kernel<<<256, 256, 0, stream>>>(st, 32 * 4096 + 2048);

  // 3) fx_mid = x @ Wfx^T + bfx ; x_mid = x @ Wx^T + bx    (bf16 out)
  dim3 gProj(M / 128, INNER / 64);
  gemm_bf16_kernel<1><<<gProj, 256, 0, stream>>>(xb, DIMC, wfxb, DIMC, bfx, fx, INNER);
  gemm_bf16_kernel<1><<<gProj, 256, 0, stream>>>(xb, DIMC, wxb, DIMC, bx, xm, INNER);

  // 4) fused slice logits + softmax + slice_norm
  slice_softmax_kernel<<<dim3(NTOK / 64, BB * HEADS), 256, 0, stream>>>(
      xm, Wslice, bslice, temp, wgt, norm);

  // 5) slice_token = w^T @ fx   (split-K WMMA, f32 atomics)
  slice_token_kernel<<<dim3(16, BB * HEADS), 256, 0, stream>>>(wgt, fx, st,
                                                               NTOK / 16);

  // 6) tiny attention over slice tokens -> out_slice (bf16)
  small_attn_kernel<<<BB * HEADS, 256, 0, stream>>>(st, norm, Wq, Wk, Wv, osl);

  // 7) unpool: out_x = w @ out_slice  (bf16, reuses x_mid buffer)
  unpool_kernel<<<dim3(NTOK / 128, BB * HEADS), 256, 0, stream>>>(osl, wgt, xm);

  // 8) final: out = out_x @ Wout^T + bout  (f32 to d_out)
  dim3 gOut(M / 128, DIMC / 64);
  gemm_bf16_kernel<0><<<gOut, 256, 0, stream>>>(xm, INNER, woutb, INNER, bout,
                                                d_out, DIMC);
}
